// Encoder_55877524521586
// MI455X (gfx1250) — matrix operand order repaired and checked by reading, not compiled, not run on hardware
//
#include <hip/hip_runtime.h>

typedef __attribute__((ext_vector_type(16))) _Float16 v16h;
typedef __attribute__((ext_vector_type(8)))  _Float16 v8h;
typedef __attribute__((ext_vector_type(4)))  _Float16 v4h;
typedef __attribute__((ext_vector_type(8)))  float    v8f;

#define NWG   16
#define BLK   256
#define SEQ   1024
#define BATCH 64
#define INP   256
#define HID   512
#define KTOT  768              // INP + HID
#define HPW   32               // hidden columns per workgroup (16 WGs * 32 = 512)
#define NPW   128              // gate columns per workgroup

#define LDS_W_STRIDE   776     // 768 + 8 f16 pad
#define LDS_IN_STRIDE  776
#define LDS_G_STRIDE   132     // 128 + 4 f32 pad
#define LDS_W_BYTES    (NPW * LDS_W_STRIDE * 2)      // 198656
#define LDS_IN_BYTES   (BATCH * LDS_IN_STRIDE * 2)   // 99328  (gate buf aliases this)
#define LDS_BYTES      (LDS_W_BYTES + LDS_IN_BYTES)  // 297984 <= 320KB/WGP

// workspace layout (bytes)
#define WS_SYNC 0
#define WS_BIAS 256
#define WS_W    8448                        // 2048*768 f16 = 3145728
#define WS_H    (WS_W + 2048 * KTOT * 2)    // 2*64*512 f16 = 131072
#define WS_X    (WS_H + 2 * BATCH * HID * 2) // 1024*64*256 f16 = 33554432

// CDNA5 async memory->LDS copy: one wave-wide instruction moves 32 lanes x 16B.
__device__ __forceinline__ void async_ld16(unsigned lds_off, const void* gaddr) {
    asm volatile("global_load_async_to_lds_b128 %0, %1, off"
                 :: "v"(lds_off), "v"((unsigned long long)(size_t)gaddr)
                 : "memory");
}
__device__ __forceinline__ void wait_async0() {
    asm volatile("s_wait_asynccnt 0x0" ::: "memory");
}

// ---------------------------------------------------------------------------
// Prep: fuse [Wx|Wh] -> f16 [2048][768], bias = bx+bh, zero h ping-pong + sync
// ---------------------------------------------------------------------------
__global__ void lstm_prep(const float* Wx0, const float* Wx1, const float* Wx2, const float* Wx3,
                          const float* Wh0, const float* Wh1, const float* Wh2, const float* Wh3,
                          const float* bx0, const float* bx1, const float* bx2, const float* bx3,
                          const float* bh0, const float* bh1, const float* bh2, const float* bh3,
                          _Float16* __restrict__ Wf, float* __restrict__ bias,
                          _Float16* __restrict__ hbuf, unsigned* __restrict__ sync) {
    const int r   = blockIdx.x;          // 0..2047: fused gate row (i,f,o,c blocks of 512)
    const int tid = threadIdx.x;
    const int g   = r >> 9;
    const int j   = r & 511;
    const float* Wxp = (g == 0) ? Wx0 : (g == 1) ? Wx1 : (g == 2) ? Wx2 : Wx3;
    const float* Whp = (g == 0) ? Wh0 : (g == 1) ? Wh1 : (g == 2) ? Wh2 : Wh3;
    for (int k = tid; k < KTOT; k += BLK) {
        float v = (k < INP) ? Wxp[j * INP + k] : Whp[j * HID + (k - INP)];
        Wf[(size_t)r * KTOT + k] = (_Float16)v;
    }
    if (tid == 0) {
        const float* bxp = (g == 0) ? bx0 : (g == 1) ? bx1 : (g == 2) ? bx2 : bx3;
        const float* bhp = (g == 0) ? bh0 : (g == 1) ? bh1 : (g == 2) ? bh2 : bh3;
        bias[r] = bxp[j] + bhp[j];
    }
    const int gtid = r * BLK + tid;
    if (gtid < (2 * BATCH * HID) / 2) ((unsigned*)hbuf)[gtid] = 0u;   // zero both h buffers
    if (gtid < 2) sync[gtid] = 0u;                                    // zero barrier state
}

// ---------------------------------------------------------------------------
// Prep: x [B][S][I] f32  ->  xf [S][B][I] f16 (contiguous per-timestep slab)
// ---------------------------------------------------------------------------
__global__ void lstm_xcvt(const float* __restrict__ x, _Float16* __restrict__ xf) {
    const long long idx = ((long long)blockIdx.x * BLK + threadIdx.x) * 4;
    if (idx >= (long long)BATCH * SEQ * INP) return;
    const int i = (int)(idx & (INP - 1));
    const long long rest = idx >> 8;
    const int t = (int)(rest & (SEQ - 1));
    const int b = (int)(rest >> 10);
    const float4 v = *(const float4*)(x + idx);
    v4h o;
    o.x = (_Float16)v.x; o.y = (_Float16)v.y; o.z = (_Float16)v.z; o.w = (_Float16)v.w;
    *(v4h*)(xf + ((size_t)t * BATCH + b) * INP + i) = o;
}

// ---------------------------------------------------------------------------
// Persistent LSTM recurrence: 16 WGs, one H-slice of 32 each; WMMA f16->f32.
// Weights live in LDS; per-step activations staged via async-to-LDS copies.
// ---------------------------------------------------------------------------
__launch_bounds__(BLK, 1)
__global__ void lstm_persistent(const _Float16* __restrict__ Wf,
                                const float* __restrict__ bias,
                                const _Float16* __restrict__ xf,
                                _Float16* __restrict__ hbuf,
                                unsigned* __restrict__ sync,
                                float* __restrict__ out) {
    extern __shared__ char smem[];
    _Float16* w_lds  = (_Float16*)smem;                    // [128][776] f16 (persistent)
    _Float16* in_lds = (_Float16*)(smem + LDS_W_BYTES);    // [64][776]  f16 : [x_t | h]
    float*    g_lds  = (float*)(smem + LDS_W_BYTES);       // [64][132]  f32 (aliases in_lds)

    const int tid   = threadIdx.x;
    const int wg    = blockIdx.x;        // 0..15
    const int wave  = tid >> 5;          // 0..7 -> N-tile (16 gate cols each)
    const int lane  = tid & 31;
    const int lhalf = lane >> 4;
    const int l16   = lane & 15;

    // ---- one-time: preload this WG's weight slice (128 gate rows) into LDS ----
    {
        const int row  = tid >> 1;                               // 0..127 local gate col
        const int half = tid & 1;
        const int grow = (row >> 5) * HID + wg * HPW + (row & 31);
        const v8h* src = (const v8h*)(Wf + (size_t)grow * KTOT + half * 384);
        v8h*       dst = (v8h*)(w_lds + row * LDS_W_STRIDE + half * 384);
#pragma unroll
        for (int i = 0; i < 48; ++i) dst[i] = src[i];
    }

    // B operand now comes from LDS: wave w covers local cols [w*16, w*16+16)
    const _Float16* bptr = w_lds + (wave * 16 + l16) * LDS_W_STRIDE + lhalf * 8;

    // A operand row bases (M-tiles over the 64 batch rows)
    const _Float16* arow0 = in_lds + (0 * 16 + l16) * LDS_IN_STRIDE + lhalf * 8;
    const _Float16* arow1 = in_lds + (1 * 16 + l16) * LDS_IN_STRIDE + lhalf * 8;
    const _Float16* arow2 = in_lds + (2 * 16 + l16) * LDS_IN_STRIDE + lhalf * 8;
    const _Float16* arow3 = in_lds + (3 * 16 + l16) * LDS_IN_STRIDE + lhalf * 8;

    // LDS byte offset of in_lds (low 32 bits of flat shared address = group offset)
    const unsigned in_off = (unsigned)(size_t)in_lds;

    // elementwise mapping: thread -> (jl hidden col, 8 batch rows)
    const int jl = tid & 31;
    const int jg = wg * HPW + jl;
    const int bb = (tid >> 5) * 8;
    const float bi = bias[0 * HID + jg];
    const float bf = bias[1 * HID + jg];
    const float bo = bias[2 * HID + jg];
    const float bc = bias[3 * HID + jg];

    float creg[8];
#pragma unroll
    for (int r = 0; r < 8; ++r) creg[r] = 0.f;

    unsigned* cnt = sync;
    unsigned* gen = sync + 1;

    __syncthreads();   // w_lds ready

    for (int t = 0; t < SEQ; ++t) {
        // ---- phase 1: async-stage [x_t | h_t] into LDS as f16 [64][776] ----
        {
            const char* xs = (const char*)(xf + (size_t)t * BATCH * INP);
#pragma unroll
            for (int i = 0; i < 8; ++i) {              // 2048 chunks of 16B
                const int e = tid + i * BLK;
                const int b = e >> 5;
                const int c = (e & 31) * 8;
                async_ld16(in_off + (unsigned)(b * LDS_IN_STRIDE + c) * 2,
                           xs + (size_t)e * 16);
            }
            const char* hs = (const char*)(hbuf + (size_t)(t & 1) * BATCH * HID);
#pragma unroll
            for (int i = 0; i < 16; ++i) {             // 4096 chunks of 16B
                const int e = tid + i * BLK;
                const int b = e >> 6;
                const int c = (e & 63) * 8;
                async_ld16(in_off + (unsigned)(b * LDS_IN_STRIDE + INP + c) * 2,
                           hs + (size_t)e * 16);
            }
            wait_async0();
        }
        __syncthreads();

        // ---- phase 2: GEMM  gates[64 x 16cols] += in[64x768] * W^T (all-LDS) ----
        v8f acc0 = {}, acc1 = {}, acc2 = {}, acc3 = {};
#pragma unroll 8
        for (int kk = 0; kk < KTOT / 32; ++kk) {
            const int k0 = kk * 32;
            union { v16h v; v8h h[2]; } bu;
            bu.h[0] = *(const v8h*)(bptr + k0);
            bu.h[1] = *(const v8h*)(bptr + k0 + 16);
            const v16h bfrag = bu.v;
            union { v16h v; v8h h[2]; } af;

            af.h[0] = *(const v8h*)(arow0 + k0);
            af.h[1] = *(const v8h*)(arow0 + k0 + 16);
            acc0 = __builtin_amdgcn_wmma_f32_16x16x32_f16(false, af.v, false, bfrag,
                                                          (short)0, acc0, false, false);
            af.h[0] = *(const v8h*)(arow1 + k0);
            af.h[1] = *(const v8h*)(arow1 + k0 + 16);
            acc1 = __builtin_amdgcn_wmma_f32_16x16x32_f16(false, af.v, false, bfrag,
                                                          (short)0, acc1, false, false);
            af.h[0] = *(const v8h*)(arow2 + k0);
            af.h[1] = *(const v8h*)(arow2 + k0 + 16);
            acc2 = __builtin_amdgcn_wmma_f32_16x16x32_f16(false, af.v, false, bfrag,
                                                          (short)0, acc2, false, false);
            af.h[0] = *(const v8h*)(arow3 + k0);
            af.h[1] = *(const v8h*)(arow3 + k0 + 16);
            acc3 = __builtin_amdgcn_wmma_f32_16x16x32_f16(false, af.v, false, bfrag,
                                                          (short)0, acc3, false, false);
        }
        __syncthreads();   // all in_lds reads done before gate buffer aliases it

        // ---- phase 3: stage accumulators (C layout: VGPR r -> M=r(+8)) ----
        {
            const int col = wave * 16 + l16;
#pragma unroll
            for (int r = 0; r < 8; ++r) {
                const int mrow = r + lhalf * 8;
                g_lds[(0 * 16 + mrow) * LDS_G_STRIDE + col] = acc0[r];
                g_lds[(1 * 16 + mrow) * LDS_G_STRIDE + col] = acc1[r];
                g_lds[(2 * 16 + mrow) * LDS_G_STRIDE + col] = acc2[r];
                g_lds[(3 * 16 + mrow) * LDS_G_STRIDE + col] = acc3[r];
            }
        }
        __syncthreads();

        // ---- phase 4: gate nonlinearity + state update ----
        {
            _Float16* hd = hbuf + (size_t)((t + 1) & 1) * BATCH * HID;
#pragma unroll
            for (int r = 0; r < 8; ++r) {
                const int b = bb + r;
                const float gi = g_lds[b * LDS_G_STRIDE +       jl] + bi;
                const float gf = g_lds[b * LDS_G_STRIDE +  32 + jl] + bf;
                const float go = g_lds[b * LDS_G_STRIDE +  64 + jl] + bo;
                const float gc = g_lds[b * LDS_G_STRIDE +  96 + jl] + bc;
                const float it = 1.f / (1.f + __expf(-gi));
                const float ft = 1.f / (1.f + __expf(-gf));
                const float ot = 1.f / (1.f + __expf(-go));
                const float gt = tanhf(gc);
                const float cn = ft * creg[r] + it * gt;
                creg[r] = cn;
                const float hn = ot * tanhf(cn);
                hd[b * HID + jg] = (_Float16)hn;
                if (t == SEQ - 1) {
                    out[b * HID + jg] = hn;                      // h_t
                    out[BATCH * HID + b * HID + jg] = cn;        // c_t
                }
            }
        }
        __syncthreads();

        // ---- device-wide barrier between timesteps (skip after last) ----
        if (t != SEQ - 1) {
            if (tid == 0) {
                __threadfence();
                unsigned prev = __hip_atomic_fetch_add(cnt, 1u, __ATOMIC_ACQ_REL,
                                                       __HIP_MEMORY_SCOPE_AGENT);
                if (prev == NWG - 1) {
                    __hip_atomic_store(cnt, 0u, __ATOMIC_RELAXED, __HIP_MEMORY_SCOPE_AGENT);
                    __hip_atomic_fetch_add(gen, 1u, __ATOMIC_RELEASE, __HIP_MEMORY_SCOPE_AGENT);
                } else {
                    const unsigned target = (unsigned)(t + 1);
                    while (__hip_atomic_load(gen, __ATOMIC_ACQUIRE,
                                             __HIP_MEMORY_SCOPE_AGENT) < target)
                        __builtin_amdgcn_s_sleep(2);
                }
            }
            __syncthreads();
        }
    }
}

// ---------------------------------------------------------------------------
extern "C" void kernel_launch(void* const* d_in, const int* in_sizes, int n_in,
                              void* d_out, int out_size, void* d_ws, size_t ws_size,
                              hipStream_t stream) {
    (void)in_sizes; (void)n_in; (void)out_size; (void)ws_size;
    const float* x = (const float*)d_in[0];
    // setup_inputs order: x, then per-gate (i,f,o,c): W_x, b_x, W_h, b_h
    const float* Wx[4] = {(const float*)d_in[1],  (const float*)d_in[5],
                          (const float*)d_in[9],  (const float*)d_in[13]};
    const float* bx[4] = {(const float*)d_in[2],  (const float*)d_in[6],
                          (const float*)d_in[10], (const float*)d_in[14]};
    const float* Wh[4] = {(const float*)d_in[3],  (const float*)d_in[7],
                          (const float*)d_in[11], (const float*)d_in[15]};
    const float* bh[4] = {(const float*)d_in[4],  (const float*)d_in[8],
                          (const float*)d_in[12], (const float*)d_in[16]};

    char* ws = (char*)d_ws;
    unsigned*  sync = (unsigned*)(ws + WS_SYNC);
    float*     bias = (float*)(ws + WS_BIAS);
    _Float16*  Wf   = (_Float16*)(ws + WS_W);
    _Float16*  hbuf = (_Float16*)(ws + WS_H);
    _Float16*  xf   = (_Float16*)(ws + WS_X);

    lstm_prep<<<2048, BLK, 0, stream>>>(Wx[0], Wx[1], Wx[2], Wx[3],
                                        Wh[0], Wh[1], Wh[2], Wh[3],
                                        bx[0], bx[1], bx[2], bx[3],
                                        bh[0], bh[1], bh[2], bh[3],
                                        Wf, bias, hbuf, sync);

    lstm_xcvt<<<(BATCH * SEQ * INP) / (4 * BLK), BLK, 0, stream>>>(x, xf);

    hipFuncSetAttribute((const void*)lstm_persistent,
                        hipFuncAttributeMaxDynamicSharedMemorySize, LDS_BYTES);
    lstm_persistent<<<NWG, BLK, LDS_BYTES, stream>>>(Wf, bias, xf, hbuf, sync,
                                                     (float*)d_out);
}
